// InteractionBlock_13254269075581
// MI455X (gfx1250) — matrix operand order repaired and checked by reading, not compiled, not run on hardware
//
#include <hip/hip_runtime.h>

#define HID 128
#define RBF 64
#define NATOMS 50000
#define NEDGES 640000
#define ETILES (NEDGES / 16)   // 40000 16-edge tiles
#define ATILES (NATOMS / 16)   // 3125 16-atom tiles (exact)

typedef __bf16 bf;
typedef __attribute__((ext_vector_type(16))) __bf16 v16bf;
typedef __attribute__((ext_vector_type(8)))  float  v8f;

static __device__ __forceinline__ v8f wmma_bf16(v16bf a, v16bf b, v8f c) {
  // D = A(16x32 bf16) * B(32x16 bf16) + C(16x16 f32)
  return __builtin_amdgcn_wmma_f32_16x16x32_bf16(false, a, false, b, (short)0, c,
                                                 false, false);
}

// Issue 8 x global_load_async_to_lds_b128: copies 4KB (16 rows x 64 f32)
// from global to an LDS staging buffer. Tracked by ASYNCcnt.
static __device__ __forceinline__ void async_tile_4k(const float* gsrc,
                                                     float* lds_dst, int lane) {
  unsigned lbase = (unsigned)(unsigned long long)lds_dst;
  unsigned long long g = (unsigned long long)(const void*)gsrc;
#pragma unroll
  for (int i = 0; i < 8; ++i) {
    unsigned byteOff = (unsigned)((i * 32 + lane) * 16);
    unsigned loff = lbase + byteOff;
    unsigned long long ga = g + byteOff;
    asm volatile("global_load_async_to_lds_b128 %0, %1, off"
                 :: "v"(loff), "v"(ga) : "memory");
  }
}

// ---------------------------------------------------------------------------
// Kernel 0: zero the aggregation workspace (harness poisons d_ws with 0xAA).
// ---------------------------------------------------------------------------
__global__ void zero_kernel(float4* __restrict__ p, int n4) {
  int i = blockIdx.x * blockDim.x + threadIdx.x;
  if (i < n4) p[i] = make_float4(0.f, 0.f, 0.f, 0.f);
}

// ---------------------------------------------------------------------------
// Kernel 1: fused edge MLP + scatter-add (persistent waves, double-buffered
// async pair-tile staging, both weight matrices resident in LDS as bf16^T).
//   em = (pair @ w_rbf + b_rbf) @ w_pair + b_pair ; agg[recv] += em
// ---------------------------------------------------------------------------
__global__ __launch_bounds__(128) void edge_kernel(
    const float* __restrict__ pair, const int* __restrict__ recv,
    const float* __restrict__ w_rbf, const float* __restrict__ b_rbf,
    const float* __restrict__ w_pair, const float* __restrict__ b_pair,
    float* __restrict__ agg) {
  __shared__ __align__(64) bf    sWr[HID * RBF];          // 16KB  w_rbf^T  [n][k]
  __shared__ __align__(64) bf    sWp[HID * HID];          // 32KB  w_pair^T [n][k]
  __shared__ __align__(64) float sStage[4][2][16 * RBF];  // 32KB  2x4KB per wave

  const int tid  = threadIdx.x;
  const int wv   = tid >> 5;
  const int lane = tid & 31;
  const int half = lane >> 4;   // 0: K-lo / C rows 0..7 ; 1: K-hi / C rows 8..15
  const int lm   = lane & 15;

  // One-time weight staging (transposed, f32 -> bf16), coalesced global reads.
  for (int i = tid; i < RBF * HID; i += 128) {
    int k = i >> 7, n = i & (HID - 1);
    sWr[n * RBF + k] = (bf)w_rbf[i];
  }
  for (int i = tid; i < HID * HID; i += 128) {
    int k = i >> 7, n = i & (HID - 1);
    sWp[n * HID + k] = (bf)w_pair[i];
  }
  // Per-lane bias registers (n = nt*16 + lm for each N-tile).
  float brbf[8], bpr[8];
#pragma unroll
  for (int nt = 0; nt < 8; ++nt) {
    brbf[nt] = b_rbf[nt * 16 + lm];
    bpr[nt]  = b_pair[nt * 16 + lm];
  }
  __syncthreads();  // only workgroup barrier in this kernel

  const int waveId = blockIdx.x * 4 + wv;
  const int nW     = gridDim.x * 4;

  int t = waveId;
  int buf = 0;
  if (t < ETILES)  // prologue: prefetch first tile
    async_tile_4k(pair + (size_t)t * 16 * RBF, &sStage[wv][0][0], lane);

  for (; t < ETILES; t += nW) {
    // Prefetch the next tile into the other buffer, then wait for current.
    int nxt = t + nW;
    if (nxt < ETILES) {
      async_tile_4k(pair + (size_t)nxt * 16 * RBF, &sStage[wv][buf ^ 1][0], lane);
      asm volatile("s_wait_asynccnt 0x8" ::: "memory");  // 8 newest may fly
    } else {
      asm volatile("s_wait_asynccnt 0x0" ::: "memory");
    }

    // A regs for GEMM1: 16x64 bf16, 2 K-steps of 32 (contiguous k = 16*half + s per K-step).
    v16bf A1[2];
    {
      const float* rp = &sStage[wv][buf][lm * RBF];
#pragma unroll
      for (int ks = 0; ks < 2; ++ks) {
        int kb = ks * 32 + 16 * half;
        v16bf a;
#pragma unroll
        for (int j = 0; j < 8; ++j) {
          a[j]     = (bf)rp[kb + j];
          a[8 + j] = (bf)rp[kb + 8 + j];
        }
        A1[ks] = a;
      }
    }

    // GEMM1: em1[16x128] = pair_tile @ w_rbf + b_rbf   (8 N-tiles)
    v8f acc[8];
#pragma unroll
    for (int nt = 0; nt < 8; ++nt) {
      v8f c;
#pragma unroll
      for (int r = 0; r < 8; ++r) c[r] = brbf[nt];
#pragma unroll
      for (int ks = 0; ks < 2; ++ks) {
        v16bf b = *(const v16bf*)&sWr[(nt * 16 + lm) * RBF + ks * 32 + 16 * half];
        c = wmma_bf16(A1[ks], b, c);
      }
      acc[nt] = c;
    }

    // Stage em1 as bf16 row-major [m][128] over the *current* buffer
    // (wave-private; the async prefetch targets the other buffer).
    bf* sE = (bf*)&sStage[wv][buf][0];
#pragma unroll
    for (int nt = 0; nt < 8; ++nt)
#pragma unroll
      for (int r = 0; r < 8; ++r)
        sE[(half * 8 + r) * HID + nt * 16 + lm] = (bf)acc[nt][r];

    // A regs for GEMM2 from staged em1 (K=128 -> 4 K-steps).
    v16bf A2[4];
    {
      const bf* rp = &sE[lm * HID];
#pragma unroll
      for (int ks = 0; ks < 4; ++ks) {
        int kb = ks * 32 + 16 * half;
        v16bf a;
#pragma unroll
        for (int j = 0; j < 8; ++j) {
          a[j]     = rp[kb + j];
          a[8 + j] = rp[kb + 8 + j];
        }
        A2[ks] = a;
      }
    }

    // Receiver indices for this tile (rows half*8 .. half*8+7).
    int rcv[8];
#pragma unroll
    for (int r = 0; r < 8; ++r) rcv[r] = recv[t * 16 + half * 8 + r];

    // GEMM2 + b_pair, scatter-add onto receiving atoms.
#pragma unroll
    for (int nt = 0; nt < 8; ++nt) {
      v8f c;
#pragma unroll
      for (int r = 0; r < 8; ++r) c[r] = bpr[nt];
#pragma unroll
      for (int ks = 0; ks < 4; ++ks) {
        v16bf b = *(const v16bf*)&sWp[(nt * 16 + lm) * HID + ks * 32 + 16 * half];
        c = wmma_bf16(A2[ks], b, c);
      }
#pragma unroll
      for (int r = 0; r < 8; ++r)
        atomicAdd(&agg[(size_t)rcv[r] * HID + nt * 16 + lm], c[r]);
    }

    buf ^= 1;
  }
}

// ---------------------------------------------------------------------------
// Kernel 2: atom MLP with residual (persistent waves, both weights in LDS):
//   out = atom_feat + silu(agg@w1+b1)@w2 + b2
// 50000 = 3125 * 16, so tiles are exact (no bounds checks).
// ---------------------------------------------------------------------------
__global__ __launch_bounds__(128) void atom_kernel(
    const float* __restrict__ atom_feat, const float* __restrict__ agg,
    const float* __restrict__ w1, const float* __restrict__ b1,
    const float* __restrict__ w2, const float* __restrict__ b2,
    float* __restrict__ out) {
  __shared__ __align__(64) bf sW1[HID * HID];   // 32KB  w1^T
  __shared__ __align__(64) bf sW2[HID * HID];   // 32KB  w2^T
  __shared__ __align__(64) bf sH[4][16 * HID];  // 16KB  per-wave silu(h) bf16

  const int tid  = threadIdx.x;
  const int wv   = tid >> 5;
  const int lane = tid & 31;
  const int half = lane >> 4;
  const int lm   = lane & 15;

  for (int i = tid; i < HID * HID; i += 128) {
    int k = i >> 7, n = i & (HID - 1);
    sW1[n * HID + k] = (bf)w1[i];
    sW2[n * HID + k] = (bf)w2[i];
  }
  float bb1[8], bb2[8];
#pragma unroll
  for (int nt = 0; nt < 8; ++nt) {
    bb1[nt] = b1[nt * 16 + lm];
    bb2[nt] = b2[nt * 16 + lm];
  }
  __syncthreads();

  const int waveId = blockIdx.x * 4 + wv;
  const int nW     = gridDim.x * 4;

  for (int t = waveId; t < ATILES; t += nW) {
    const int atom0 = t * 16;

    // A regs from agg (global f32 -> bf16), row = atom0 + lm.
    const float* rp = agg + (size_t)(atom0 + lm) * HID;
    v16bf A[4];
#pragma unroll
    for (int ks = 0; ks < 4; ++ks) {
      int kb = ks * 32 + 16 * half;
      v16bf a;
#pragma unroll
      for (int j = 0; j < 8; ++j) {
        a[j]     = (bf)rp[kb + j];
        a[8 + j] = (bf)rp[kb + 8 + j];
      }
      A[ks] = a;
    }

    // GEMM1 + bias + silu, stage bf16 into per-wave region.
#pragma unroll
    for (int nt = 0; nt < 8; ++nt) {
      v8f c;
#pragma unroll
      for (int r = 0; r < 8; ++r) c[r] = bb1[nt];
#pragma unroll
      for (int ks = 0; ks < 4; ++ks) {
        v16bf b = *(const v16bf*)&sW1[(nt * 16 + lm) * HID + ks * 32 + 16 * half];
        c = wmma_bf16(A[ks], b, c);
      }
#pragma unroll
      for (int r = 0; r < 8; ++r) {
        float x = c[r];
        float s = x / (1.0f + __expf(-x));  // silu
        sH[wv][(half * 8 + r) * HID + nt * 16 + lm] = (bf)s;
      }
    }

    // A2 from staged h.
    v16bf A2[4];
    {
      const bf* hp = &sH[wv][lm * HID];
#pragma unroll
      for (int ks = 0; ks < 4; ++ks) {
        int kb = ks * 32 + 16 * half;
        v16bf a;
#pragma unroll
        for (int j = 0; j < 8; ++j) {
          a[j]     = hp[kb + j];
          a[8 + j] = hp[kb + 8 + j];
        }
        A2[ks] = a;
      }
    }

    // GEMM2 + bias + residual -> out.
#pragma unroll
    for (int nt = 0; nt < 8; ++nt) {
      v8f c;
#pragma unroll
      for (int r = 0; r < 8; ++r) c[r] = bb2[nt];
#pragma unroll
      for (int ks = 0; ks < 4; ++ks) {
        v16bf b = *(const v16bf*)&sW2[(nt * 16 + lm) * HID + ks * 32 + 16 * half];
        c = wmma_bf16(A2[ks], b, c);
      }
#pragma unroll
      for (int r = 0; r < 8; ++r) {
        size_t idx = (size_t)(atom0 + half * 8 + r) * HID + nt * 16 + lm;
        out[idx] = atom_feat[idx] + c[r];
      }
    }
  }
}

// ---------------------------------------------------------------------------
extern "C" void kernel_launch(void* const* d_in, const int* in_sizes, int n_in,
                              void* d_out, int out_size, void* d_ws,
                              size_t ws_size, hipStream_t stream) {
  const float* atom_feat = (const float*)d_in[0];
  const float* pair      = (const float*)d_in[1];
  const int*   recv      = (const int*)d_in[2];
  const float* w_rbf     = (const float*)d_in[3];
  const float* b_rbf     = (const float*)d_in[4];
  const float* w_pair    = (const float*)d_in[5];
  const float* b_pair    = (const float*)d_in[6];
  const float* w1        = (const float*)d_in[7];
  const float* b1        = (const float*)d_in[8];
  const float* w2        = (const float*)d_in[9];
  const float* b2        = (const float*)d_in[10];
  float* out = (float*)d_out;
  float* agg = (float*)d_ws;  // [NATOMS, HID] f32 aggregation buffer

  int n4 = NATOMS * HID / 4;
  zero_kernel<<<dim3((n4 + 255) / 256), dim3(256), 0, stream>>>((float4*)agg, n4);
  // Persistent grids: 1000 blocks x 4 waves -> 10 edge tiles per wave;
  // 200 blocks x 4 waves -> ~4 atom tiles per wave.
  edge_kernel<<<dim3(1000), dim3(128), 0, stream>>>(
      pair, recv, w_rbf, b_rbf, w_pair, b_pair, agg);
  atom_kernel<<<dim3(200), dim3(128), 0, stream>>>(
      atom_feat, agg, w1, b1, w2, b2, out);
}
